// LinkEncoder_89069031784547
// MI455X (gfx1250) — matrix operand order repaired and checked by reading, not compiled, not run on hardware
//
#include <hip/hip_runtime.h>
#include <math.h>

// ---------------- Types for WMMA (gfx1250, wave32) ----------------
typedef __attribute__((ext_vector_type(16))) __bf16 v16bf;
typedef __attribute__((ext_vector_type(8)))  float  v8f;
typedef __attribute__((ext_vector_type(4)))  unsigned int uv4;

union FragBF { v16bf v; uv4 q[2]; unsigned short u[16]; };

__device__ __forceinline__ unsigned short f2bf(float x) {
  unsigned int u = __float_as_uint(x);
  unsigned int r = (u + 0x7FFFu + ((u >> 16) & 1u)) >> 16;
  return (unsigned short)r;
}
__device__ __forceinline__ float bf2f(unsigned short b) {
  return __uint_as_float(((unsigned int)b) << 16);
}
__device__ __forceinline__ float gelu_exact(float x) {
  return 0.5f * x * (1.0f + erff(x * 0.70710678118654752f));
}

// Problem dims (fixed by reference)
#define H_DIM   100   // hidden
#define K_TOK   20    // tokens per node
#define T_CH    100   // time channels
#define IN_F    64    // edge attr features
#define OUT_D   100
#define KTH     192   // 164 padded to 6*32
#define KH      128   // 100 padded to 4*32
#define C4      400   // 4*H
#define K2      416   // 400 padded to 13*32
#define HP      112   // 100 padded to 7*16
#define EB      64    // edges per block in edge kernel

// WMMA k-step chunk: preload NK fragments of A (LDS/row-major, stride already applied by caller)
// and B (row-major, stride already applied by caller), then issue NK WMMAs.
// arow must be pre-offset by hi*8, brow by hi*8 (bf16 A/B lane layouts).
template <int NK>
__device__ __forceinline__ void wmma_chunk(v8f& acc, const unsigned short* arow,
                                           const unsigned short* brow, int c) {
  FragBF afr[NK], bfr[NK];
#pragma unroll
  for (int j = 0; j < NK; ++j) {
    int k0 = (c + j) * 32;
    afr[j].q[0] = *(const uv4*)(arow + k0);
    afr[j].q[1] = *(const uv4*)(arow + k0 + 16);
    bfr[j].q[0] = *(const uv4*)(brow + k0);
    bfr[j].q[1] = *(const uv4*)(brow + k0 + 16);
  }
#pragma unroll
  for (int j = 0; j < NK; ++j)
    acc = __builtin_amdgcn_wmma_f32_16x16x32_bf16(false, afr[j].v, false, bfr[j].v,
                                                  (short)0, acc, false, false);
}

// ================= Kernel 1: weight conversion + counter zero =================
__global__ void prep_kernel(const float* thW, const float* cl1W, const float* cl2W,
                            unsigned short* Wth, unsigned short* Wcl1, unsigned short* Wcl2,
                            int* deg, int* cursor, int n) {
  int i0 = blockIdx.x * 256 + threadIdx.x;
  int stride = gridDim.x * 256;
  for (int i = i0; i < HP * KTH; i += stride) {      // th_W [100][164] -> [112][192]
    int r = i / KTH, c = i - r * KTH;
    Wth[i] = (r < H_DIM && c < (T_CH + IN_F)) ? f2bf(thW[r * (T_CH + IN_F) + c]) : (unsigned short)0;
  }
  for (int i = i0; i < C4 * KH; i += stride) {       // cl1_W [400][100] -> [400][128]
    int r = i >> 7, c = i & 127;
    Wcl1[i] = (c < H_DIM) ? f2bf(cl1W[r * H_DIM + c]) : (unsigned short)0;
  }
  for (int i = i0; i < HP * K2; i += stride) {       // cl2_W [100][400] -> [112][416]
    int r = i / K2, c = i - r * K2;
    Wcl2[i] = (r < H_DIM && c < C4) ? f2bf(cl2W[r * C4 + c]) : (unsigned short)0;
  }
  for (int i = i0; i < n; i += stride) { deg[i] = 0; cursor[i] = 0; }
}

// ================= Kernel 2: edge featurization via WMMA =================
// 64 edges/block (4 M-tiles), 7 waves; wave w owns N-tile w across all 4 M-tiles.
// B fragments (weights) preloaded for all 6 k-steps -> 4x reuse per B load.
__global__ __launch_bounds__(224) void edge_feat_kernel(
    const int* ei, const float* eattr, const float* etime,
    const float* stime, const unsigned short* Wth,
    const float* thb, unsigned short* feat, int E) {
  __shared__ unsigned short As[EB * KTH];     // 24 KB
  __shared__ float st_s[EB], et_s[EB];
  const int tid = threadIdx.x;                // 224 threads = 7 waves
  const int e0 = blockIdx.x * EB;

  for (int i = tid; i < EB; i += 224) {
    int e = e0 + i;
    if (e < E) { int col = ei[E + e]; st_s[i] = stime[col]; et_s[i] = etime[e]; }
    else       { st_s[i] = 0.f; et_s[i] = 0.f; }
  }
  __syncthreads();
  // stage A tile [64 edges][192] bf16 in LDS (time enc on the fly)
  for (int idx = tid; idx < EB * KTH; idx += 224) {
    int r = idx / KTH, k = idx - r * KTH;
    int e = e0 + r;
    float v = 0.f;
    if (e < E) {
      if (k < T_CH) {
        float dt = st_s[r] - et_s[r];
        float w = exp2f(-(float)k * 0.33219280948873623f); // 10^(-k/10)
        v = cosf(dt * w);
      } else if (k < T_CH + IN_F) {
        v = eattr[(size_t)e * IN_F + (k - T_CH)];
      }
    }
    As[idx] = f2bf(v);
  }
  __syncthreads();

  const int wave = tid >> 5, lane = tid & 31;       // wave = N-tile 0..6
  const int hi = (lane >= 16) ? 1 : 0, lrow = lane & 15;
  const int nglob = wave * 16 + lrow;
  const float bias = (nglob < H_DIM) ? thb[nglob] : 0.f;

  v8f acc[4];
#pragma unroll
  for (int mt = 0; mt < 4; ++mt)
#pragma unroll
    for (int r = 0; r < 8; ++r) acc[mt][r] = bias;

  // preload all 6 B fragments (12 x b128 -> one clause)
  FragBF bfr[6];
  const unsigned short* brow = Wth + nglob * KTH + hi * 8;
#pragma unroll
  for (int ks = 0; ks < 6; ++ks) {
    bfr[ks].q[0] = *(const uv4*)(brow + ks * 32);
    bfr[ks].q[1] = *(const uv4*)(brow + ks * 32 + 16);
  }
#pragma unroll
  for (int ks = 0; ks < 6; ++ks) {
    int k0 = ks * 32;
#pragma unroll
    for (int mt = 0; mt < 4; ++mt) {
      FragBF a;
      const unsigned short* ap = As + (mt * 16 + lrow) * KTH + k0 + hi * 8;
      a.q[0] = *(const uv4*)ap;
      a.q[1] = *(const uv4*)(ap + 16);
      acc[mt] = __builtin_amdgcn_wmma_f32_16x16x32_bf16(false, a.v, false, bfr[ks].v,
                                                        (short)0, acc[mt], false, false);
    }
  }
#pragma unroll
  for (int mt = 0; mt < 4; ++mt) {
#pragma unroll
    for (int r = 0; r < 8; ++r) {
      int m = mt * 16 + r + hi * 8;
      int e = e0 + m;
      if (e < E && nglob < H_DIM) feat[(size_t)e * H_DIM + nglob] = f2bf(acc[mt][r]);
    }
  }
}

// ================= Kernel 3: per-destination valid-edge counts =================
__global__ void count_kernel(const int* ei, const float* etime, const float* stime,
                             int* deg, int E) {
  int e = blockIdx.x * 256 + threadIdx.x;
  if (e >= E) return;
  int col = ei[E + e];
  if (etime[e] <= stime[col]) atomicAdd(&deg[col], 1);
}

// ================= Kernel 4: exclusive scan (single block) =================
__global__ void scan_kernel(const int* deg, int* off, int n) {
  __shared__ int buf[1024];
  __shared__ int total;
  if (threadIdx.x == 0) { total = 0; off[0] = 0; }
  __syncthreads();
  for (int base = 0; base < n; base += 1024) {
    int i = base + threadIdx.x;
    buf[threadIdx.x] = (i < n) ? deg[i] : 0;
    __syncthreads();
    for (int s = 1; s < 1024; s <<= 1) {
      int t = (threadIdx.x >= (unsigned)s) ? buf[threadIdx.x - s] : 0;
      __syncthreads();
      buf[threadIdx.x] += t;
      __syncthreads();
    }
    if (i < n) off[i + 1] = total + buf[threadIdx.x];
    __syncthreads();
    if (threadIdx.x == 0) total += buf[1023];
    __syncthreads();
  }
}

// ================= Kernel 5: scatter edge ids into CSR =================
__global__ void scatter_kernel(const int* ei, const float* etime, const float* stime,
                               const int* off, int* cursor, int* eids, int E) {
  int e = blockIdx.x * 256 + threadIdx.x;
  if (e >= E) return;
  int col = ei[E + e];
  if (etime[e] <= stime[col]) {
    int p = atomicAdd(&cursor[col], 1);
    eids[off[col] + p] = e;
  }
}

// ================= Kernel 6: per-node top-K by time desc (tie: smaller id) ===
__global__ void topk_kernel(const int* off, const int* eids, const float* etime,
                            int* sel, int n) {
  int node = blockIdx.x * 128 + threadIdx.x;
  if (node >= n) return;
  int s = off[node], e = off[node + 1];
  float tt[K_TOK]; int id[K_TOK]; int cnt = 0;
  for (int j = s; j < e; ++j) {
    int eid = eids[j];
    float t = etime[eid];
    if (cnt < K_TOK) {
      int p = cnt;
      while (p > 0 && (tt[p-1] < t || (tt[p-1] == t && id[p-1] > eid))) {
        tt[p] = tt[p-1]; id[p] = id[p-1]; --p;
      }
      tt[p] = t; id[p] = eid; ++cnt;
    } else if (t > tt[K_TOK-1] || (t == tt[K_TOK-1] && eid < id[K_TOK-1])) {
      int p = K_TOK - 1;
      while (p > 0 && (tt[p-1] < t || (tt[p-1] == t && id[p-1] > eid))) {
        tt[p] = tt[p-1]; id[p] = id[p-1]; --p;
      }
      tt[p] = t; id[p] = eid;
    }
  }
  for (int k = 0; k < K_TOK; ++k) sel[(size_t)node * K_TOK + k] = (k < cnt) ? id[k] : -1;
}

// ================= Kernel 7: per-node MLP-Mixer (WMMA channel mix) =========
struct MixerArgs {
  const unsigned short* feat;
  const int* sel;
  const float *tng, *tnb, *tl1W, *tl1b, *tl2W, *tl2b;
  const float *cng, *cnb, *cl1b, *cl2b, *hng, *hnb, *hlW, *hlb;
  const unsigned short *Wcl1, *Wcl2;
  float* out;
};

__global__ __launch_bounds__(256) void mixer_kernel(MixerArgs A) {
  __shared__ unsigned short xs[32 * KH];      // bf16 staging (x, then LN(h_tok))
  __shared__ unsigned short gbuf[32 * K2];    // gelu(GEMM1) bf16
  __shared__ float xf[K_TOK * H_DIM];         // x fp32
  __shared__ float lnbuf[K_TOK * H_DIM];      // LN scratch / final LN
  __shared__ float htok[K_TOK * H_DIM];       // token-mix output, then h_ch
  __shared__ float mvec[H_DIM];
  __shared__ int   sels[K_TOK];

  const int tid = threadIdx.x;                // 256 threads = 8 waves
  const int node = blockIdx.x;
  const int wave = tid >> 5, lane = tid & 31;
  const int hi = (lane >= 16) ? 1 : 0, lrow = lane & 15;

  // warm channel-mix weights into cache (global_prefetch)
  __builtin_prefetch(A.Wcl1 + (size_t)tid * 128, 0, 1);
  __builtin_prefetch(A.Wcl2 + (size_t)tid * 128, 0, 1);

  // zero pads
  for (int i = tid; i < 32 * KH; i += 256) xs[i] = 0;
  for (int i = tid; i < 32 * K2; i += 256) gbuf[i] = 0;
  if (tid < K_TOK) sels[tid] = A.sel[(size_t)node * K_TOK + tid];
  __syncthreads();

  // gather x
  for (int idx = tid; idx < K_TOK * H_DIM; idx += 256) {
    int k = idx / H_DIM, h = idx - k * H_DIM;
    int e = sels[k];
    float v = 0.f;
    if (e >= 0) {
      unsigned short us = A.feat[(size_t)e * H_DIM + h];
      xs[k * KH + h] = us;
      v = bf2f(us);
    }
    xf[idx] = v;
  }
  __syncthreads();

  // token mixing: LN over H per token
  if (tid < K_TOK) {
    const float* row = xf + tid * H_DIM;
    float mu = 0.f;
    for (int h = 0; h < H_DIM; ++h) mu += row[h];
    mu *= (1.0f / H_DIM);
    float var = 0.f;
    for (int h = 0; h < H_DIM; ++h) { float d = row[h] - mu; var += d * d; }
    var *= (1.0f / H_DIM);
    float rs = rsqrtf(var + 1e-5f);
    for (int h = 0; h < H_DIM; ++h)
      lnbuf[tid * H_DIM + h] = (row[h] - mu) * rs * A.tng[h] + A.tnb[h];
  }
  __syncthreads();
  // small transposed matmuls over token dim (per h-channel thread)
  if (tid < H_DIM) {
    int h = tid;
    float u[K_TOK / 2];
#pragma unroll
    for (int j = 0; j < K_TOK / 2; ++j) {
      float s = A.tl1b[j];
      for (int k = 0; k < K_TOK; ++k) s += lnbuf[k * H_DIM + h] * A.tl1W[j * K_TOK + k];
      u[j] = gelu_exact(s);
    }
    for (int k = 0; k < K_TOK; ++k) {
      float s = A.tl2b[k];
#pragma unroll
      for (int j = 0; j < K_TOK / 2; ++j) s += u[j] * A.tl2W[k * (K_TOK / 2) + j];
      htok[k * H_DIM + h] = s + xf[k * H_DIM + h];
    }
  }
  __syncthreads();

  // channel mixing: LN(h_tok) -> bf16 into xs
  if (tid < K_TOK) {
    const float* row = htok + tid * H_DIM;
    float mu = 0.f;
    for (int h = 0; h < H_DIM; ++h) mu += row[h];
    mu *= (1.0f / H_DIM);
    float var = 0.f;
    for (int h = 0; h < H_DIM; ++h) { float d = row[h] - mu; var += d * d; }
    var *= (1.0f / H_DIM);
    float rs = rsqrtf(var + 1e-5f);
    for (int h = 0; h < H_DIM; ++h)
      xs[tid * KH + h] = f2bf((row[h] - mu) * rs * A.cng[h] + A.cnb[h]);
  }
  __syncthreads();

  // GEMM1: [32x128] x [128->400] + bias, gelu -> gbuf bf16 (4 k-steps preloaded)
  for (int p = wave; p < 50; p += 8) {
    int mt = p / 25, nt = p - mt * 25;
    int nglob = nt * 16 + lrow;
    float bias = A.cl1b[nglob];
    v8f acc;
#pragma unroll
    for (int r = 0; r < 8; ++r) acc[r] = bias;
    const unsigned short* arow = xs + (mt * 16 + lrow) * KH + hi * 8;
    const unsigned short* brow = A.Wcl1 + nglob * KH + hi * 8;
    wmma_chunk<4>(acc, arow, brow, 0);
#pragma unroll
    for (int r = 0; r < 8; ++r) {
      int m = mt * 16 + r + hi * 8;
      gbuf[m * K2 + nglob] = f2bf(gelu_exact(acc[r]));
    }
  }
  __syncthreads();

  // GEMM2: [32x416] x [416->112] + bias, residual into htok (13 k-steps, chunks)
  for (int p = wave; p < 14; p += 8) {
    int mt = p / 7, nt = p - mt * 7;
    int nglob = nt * 16 + lrow;
    float bias = (nglob < H_DIM) ? A.cl2b[nglob] : 0.f;
    v8f acc;
#pragma unroll
    for (int r = 0; r < 8; ++r) acc[r] = bias;
    const unsigned short* arow = gbuf + (mt * 16 + lrow) * K2 + hi * 8;
    const unsigned short* brow = A.Wcl2 + nglob * K2 + hi * 8;
    wmma_chunk<4>(acc, arow, brow, 0);
    wmma_chunk<4>(acc, arow, brow, 4);
    wmma_chunk<4>(acc, arow, brow, 8);
    wmma_chunk<1>(acc, arow, brow, 12);
#pragma unroll
    for (int r = 0; r < 8; ++r) {
      int m = mt * 16 + r + hi * 8;
      if (m < K_TOK && nglob < H_DIM) htok[m * H_DIM + nglob] += acc[r];
    }
  }
  __syncthreads();

  // final LN
  if (tid < K_TOK) {
    const float* row = htok + tid * H_DIM;
    float mu = 0.f;
    for (int h = 0; h < H_DIM; ++h) mu += row[h];
    mu *= (1.0f / H_DIM);
    float var = 0.f;
    for (int h = 0; h < H_DIM; ++h) { float d = row[h] - mu; var += d * d; }
    var *= (1.0f / H_DIM);
    float rs = rsqrtf(var + 1e-5f);
    for (int h = 0; h < H_DIM; ++h)
      lnbuf[tid * H_DIM + h] = (row[h] - mu) * rs * A.hng[h] + A.hnb[h];
  }
  __syncthreads();
  if (tid < H_DIM) {
    float s = 0.f;
    for (int k = 0; k < K_TOK; ++k) s += lnbuf[k * H_DIM + tid];
    mvec[tid] = s * (1.0f / K_TOK);
  }
  __syncthreads();
  if (tid < OUT_D) {
    float s = A.hlb[tid];
    for (int h = 0; h < H_DIM; ++h) s += mvec[h] * A.hlW[tid * H_DIM + h];
    A.out[(size_t)node * OUT_D + tid] = s;
  }
}

// ================================ Launch ================================
extern "C" void kernel_launch(void* const* d_in, const int* in_sizes, int n_in,
                              void* d_out, int out_size, void* d_ws, size_t ws_size,
                              hipStream_t stream) {
  const int*   edge_index = (const int*)d_in[0];
  const float* edge_attr  = (const float*)d_in[1];
  const float* edge_time  = (const float*)d_in[2];
  const float* seed_time  = (const float*)d_in[3];
  const float* th_W  = (const float*)d_in[4];
  const float* th_b  = (const float*)d_in[5];
  const float* tn_g  = (const float*)d_in[6];
  const float* tn_b  = (const float*)d_in[7];
  const float* tl1_W = (const float*)d_in[8];
  const float* tl1_b = (const float*)d_in[9];
  const float* tl2_W = (const float*)d_in[10];
  const float* tl2_b = (const float*)d_in[11];
  const float* cn_g  = (const float*)d_in[12];
  const float* cn_b  = (const float*)d_in[13];
  const float* cl1_W = (const float*)d_in[14];
  const float* cl1_b = (const float*)d_in[15];
  const float* cl2_W = (const float*)d_in[16];
  const float* cl2_b = (const float*)d_in[17];
  const float* hn_g  = (const float*)d_in[18];
  const float* hn_b  = (const float*)d_in[19];
  const float* hl_W  = (const float*)d_in[20];
  const float* hl_b  = (const float*)d_in[21];
  const int E = in_sizes[2];
  const int N = in_sizes[3];

  // workspace carve-up (256B aligned)
  char* wsb = (char*)d_ws;
  size_t off_b = 0;
  auto carve = [&](size_t bytes) -> char* {
    char* p = wsb + off_b;
    off_b = (off_b + bytes + 255) & ~(size_t)255;
    return p;
  };
  unsigned short* feat = (unsigned short*)carve((size_t)E * H_DIM * 2);
  unsigned short* Wth  = (unsigned short*)carve((size_t)HP * KTH * 2);
  unsigned short* Wcl1 = (unsigned short*)carve((size_t)C4 * KH * 2);
  unsigned short* Wcl2 = (unsigned short*)carve((size_t)HP * K2 * 2);
  int* deg    = (int*)carve((size_t)N * 4);
  int* offs   = (int*)carve((size_t)(N + 1) * 4);
  int* cursor = (int*)carve((size_t)N * 4);
  int* eids   = (int*)carve((size_t)E * 4);
  int* sel    = (int*)carve((size_t)N * K_TOK * 4);
  (void)ws_size; (void)n_in; (void)out_size;

  prep_kernel<<<256, 256, 0, stream>>>(th_W, cl1_W, cl2_W, Wth, Wcl1, Wcl2, deg, cursor, N);
  edge_feat_kernel<<<(E + EB - 1) / EB, 224, 0, stream>>>(edge_index, edge_attr, edge_time,
                                                          seed_time, Wth, th_b, feat, E);
  count_kernel<<<(E + 255) / 256, 256, 0, stream>>>(edge_index, edge_time, seed_time, deg, E);
  scan_kernel<<<1, 1024, 0, stream>>>(deg, offs, N);
  scatter_kernel<<<(E + 255) / 256, 256, 0, stream>>>(edge_index, edge_time, seed_time,
                                                      offs, cursor, eids, E);
  topk_kernel<<<(N + 127) / 128, 128, 0, stream>>>(offs, eids, edge_time, sel, N);

  MixerArgs ma;
  ma.feat = feat; ma.sel = sel;
  ma.tng = tn_g; ma.tnb = tn_b; ma.tl1W = tl1_W; ma.tl1b = tl1_b;
  ma.tl2W = tl2_W; ma.tl2b = tl2_b; ma.cng = cn_g; ma.cnb = cn_b;
  ma.cl1b = cl1_b; ma.cl2b = cl2_b; ma.hng = hn_g; ma.hnb = hn_b;
  ma.hlW = hl_W; ma.hlb = hl_b; ma.Wcl1 = Wcl1; ma.Wcl2 = Wcl2;
  ma.out = (float*)d_out;
  mixer_kernel<<<N, 256, 0, stream>>>(ma);
}
